// CUBASpikingCNN_8847632629952
// MI455X (gfx1250) — matrix operand order repaired and checked by reading, not compiled, not run on hardware
//
#include <hip/hip_runtime.h>
#include <hip/hip_bf16.h>

typedef __attribute__((ext_vector_type(16))) _Float16 v16h;
typedef __attribute__((ext_vector_type(8)))  _Float16 v8h;
typedef __attribute__((ext_vector_type(8)))  float    v8f;
typedef __attribute__((ext_vector_type(4)))  unsigned int u32x4;
typedef __attribute__((ext_vector_type(4)))  int      i32x4;
typedef __attribute__((ext_vector_type(8)))  int      i32x8;

union V16 { v16h v; v8h h[2]; };

#define WMMA16(acc, A, B)                                                     \
  (acc) = __builtin_amdgcn_wmma_f32_16x16x32_f16(false, (A), false, (B),      \
                                                 (short)0, (acc), false, false)

// ---------------- workspace layout (bytes) ----------------
constexpr size_t OFF_W2T   = 0;         // f16 [9*64][128]   conv2 weights, k-major
constexpr size_t OFF_W3T   = 147456;    // f16 [9*128][256]
constexpr size_t OFF_WTCT  = 737280;    // f16 [3*256][256]  temporal conv W'[k*256+c][d]
constexpr size_t OFF_WRECT = 1130496;   // f16 [256][256]    rec_w^T
constexpr size_t OFF_WF1T  = 1261568;   // f16 [256][128]    fc1_w^T
constexpr size_t OFF_S1    = 1327104;   // f16 [2048][64]    layer1 spikes NHWC (GEMM A)
constexpr size_t OFF_S1S   = 1589248;   // f16 swizzled spikes (LIF-private)
constexpr size_t OFF_CUR1S = 1851392;   // f32 swizzled
constexpr size_t OFF_VOL1S = 2375680;   // f32 swizzled
constexpr size_t OFF_S2    = 2899968;   // f16 [1152][128]   layer2 spikes NHWC
constexpr size_t OFF_S2S   = 3194880;   // f16 swizzled
constexpr size_t OFF_CUR2S = 3489792;   // f32 swizzled
constexpr size_t OFF_VOL2S = 4079616;   // f32 swizzled
constexpr size_t OFF_P2    = 4669440;   // f16 [32*9][128]   pooled NHWC
constexpr size_t OFF_CUR3  = 4743168;   // f32 [32][256]
constexpr size_t OFF_VOL3  = 4775936;
constexpr size_t OFF_SP3   = 4808704;   // f16 [32][256]
constexpr size_t OFF_INS   = 4825088;   // f16 [3][32][256] ring
constexpr size_t OFF_CUR4  = 4874240;
constexpr size_t OFF_VOL4  = 4907008;
constexpr size_t OFF_SP4   = 4939776;
constexpr size_t OFF_CUR5  = 4956160;
constexpr size_t OFF_VOL5  = 4988928;
constexpr size_t OFF_SP5   = 5021696;
constexpr size_t OFF_CUR6  = 5038080;   // f32 [32][128]
constexpr size_t OFF_VOL6  = 5054464;
constexpr size_t OFF_SP6   = 5070848;   // f16 [32][128]

__device__ __forceinline__ float lif_step(float psp, float* __restrict__ cur,
                                          float* __restrict__ vol,
                                          _Float16* __restrict__ sp) {
  float c = __builtin_fmaf(0.5f, *cur, psp);
  float sprev = (float)(*sp);
  float v = __builtin_fmaf(0.75f * (1.0f - sprev), *vol, c);
  float s = (v > 0.5f) ? 1.0f : 0.0f;
  *cur = c; *vol = v; *sp = (_Float16)s;
  return s;
}

__device__ __forceinline__ v16h ld_pair(const _Float16* p, int gap) {
  V16 t;
  t.h[0] = *(const v8h*)p;
  t.h[1] = *(const v8h*)(p + gap);
  return t.v;
}

extern "C" __global__ void __launch_bounds__(1024)
snn_fused(const float* __restrict__ x,
          const float* __restrict__ w1,  const float* __restrict__ b1c,
          const float* __restrict__ w2,  const float* __restrict__ b2c,
          const float* __restrict__ w3,  const float* __restrict__ b3c,
          const float* __restrict__ tcw, const float* __restrict__ tcbc,
          const float* __restrict__ recw,const float* __restrict__ recbc,
          const float* __restrict__ f1w, const float* __restrict__ f1bc,
          const float* __restrict__ f2w, const float* __restrict__ tswp,
          const float* __restrict__ maskfc,
          const float* __restrict__ c1s, const float* __restrict__ c2s,
          const float* __restrict__ c3s, const float* __restrict__ tcs,
          const float* __restrict__ rsin,const float* __restrict__ fsin,
          float* __restrict__ out, char* __restrict__ ws) {
  const int tid   = threadIdx.x;
  const int wv    = tid >> 5;
  const int lane  = tid & 31;
  const int laneM = lane & 15;
  const int h8    = (lane >> 4) << 3;

  _Float16* __restrict__ w2t   = (_Float16*)(ws + OFF_W2T);
  _Float16* __restrict__ w3t   = (_Float16*)(ws + OFF_W3T);
  _Float16* __restrict__ wtct  = (_Float16*)(ws + OFF_WTCT);
  _Float16* __restrict__ wrect = (_Float16*)(ws + OFF_WRECT);
  _Float16* __restrict__ wf1t  = (_Float16*)(ws + OFF_WF1T);
  _Float16* __restrict__ s1    = (_Float16*)(ws + OFF_S1);
  _Float16* __restrict__ s1s   = (_Float16*)(ws + OFF_S1S);
  float*    __restrict__ cur1s = (float*)(ws + OFF_CUR1S);
  float*    __restrict__ vol1s = (float*)(ws + OFF_VOL1S);
  _Float16* __restrict__ s2    = (_Float16*)(ws + OFF_S2);
  _Float16* __restrict__ s2s   = (_Float16*)(ws + OFF_S2S);
  float*    __restrict__ cur2s = (float*)(ws + OFF_CUR2S);
  float*    __restrict__ vol2s = (float*)(ws + OFF_VOL2S);
  _Float16* __restrict__ p2    = (_Float16*)(ws + OFF_P2);
  float*    __restrict__ cur3  = (float*)(ws + OFF_CUR3);
  float*    __restrict__ vol3  = (float*)(ws + OFF_VOL3);
  _Float16* __restrict__ sp3   = (_Float16*)(ws + OFF_SP3);
  _Float16* __restrict__ ins   = (_Float16*)(ws + OFF_INS);
  float*    __restrict__ cur4  = (float*)(ws + OFF_CUR4);
  float*    __restrict__ vol4  = (float*)(ws + OFF_VOL4);
  _Float16* __restrict__ sp4   = (_Float16*)(ws + OFF_SP4);
  float*    __restrict__ cur5  = (float*)(ws + OFF_CUR5);
  float*    __restrict__ vol5  = (float*)(ws + OFF_VOL5);
  _Float16* __restrict__ sp5   = (_Float16*)(ws + OFF_SP5);
  float*    __restrict__ cur6  = (float*)(ws + OFF_CUR6);
  float*    __restrict__ vol6  = (float*)(ws + OFF_VOL6);
  _Float16* __restrict__ sp6   = (_Float16*)(ws + OFF_SP6);

  __shared__ __align__(16) float    lds_xstage[3200 * 4]; // TDM landing pad (4 timesteps)
  __shared__ __align__(16) _Float16 lds_x[3200];          // x_t  [32][10][10] f16
  __shared__ __align__(16) _Float16 lds_w1t[2048];        // w1^T [32(k,pad)][64]
  __shared__ float lds_b1[64];
  __shared__ float lds_b2[128];
  __shared__ float lds_b3[256];
  __shared__ float lds_btc[768];
  __shared__ float lds_brec[256];
  __shared__ float lds_bf1[128];
  __shared__ float lds_fc2w[256];
  __shared__ float lds_acc[64];

  // ---------------- init: weight transposes (f32->f16) + state import ----
  for (int i = tid; i < 73728; i += 1024) {       // w2t[(khw*64+c)][n]
    int n = i & 127, r = i >> 7, khw = r >> 6, c = r & 63;
    w2t[i] = (_Float16)w2[(n * 64 + c) * 9 + khw];
  }
  for (int i = tid; i < 294912; i += 1024) {      // w3t[(khw*128+c)][n]
    int n = i & 255, r = i >> 8, khw = r >> 7, c = r & 127;
    w3t[i] = (_Float16)w3[(n * 128 + c) * 9 + khw];
  }
  for (int i = tid; i < 196608; i += 1024) {      // wtct[(k*256+c)][d]
    int d = i & 255, r = i >> 8, k = r >> 8, c = r & 255;
    wtct[i] = (_Float16)tcw[(k * 256 + d) * 256 + c];
  }
  for (int i = tid; i < 65536; i += 1024) {       // wrect[c][d]
    int d = i & 255, c = i >> 8;
    wrect[i] = (_Float16)recw[d * 256 + c];
  }
  for (int i = tid; i < 32768; i += 1024) {       // wf1t[c][n]
    int n = i & 127, c = i >> 7;
    wf1t[i] = (_Float16)f1w[n * 256 + c];
  }
  // layer1 state: NCHW input -> WMMA-swizzled [(tile*32+lane)*8+r]
  for (int i = tid; i < 131072; i += 1024) {
    int r = i & 7, ln = (i >> 3) & 31, tile = i >> 8;
    int rt = tile >> 2, ct = tile & 3;
    int orow = rt * 16 + ((ln >> 4) << 3) + r;
    int col  = ct * 16 + (ln & 15);
    int b = orow >> 6, pix = orow & 63;
    int src = (b * 64 + col) * 64 + pix;
    s1s[i]   = (_Float16)c1s[src];
    cur1s[i] = c1s[131072 + src];
    vol1s[i] = c1s[262144 + src];
  }
  // layer2 state
  for (int i = tid; i < 147456; i += 1024) {
    int r = i & 7, ln = (i >> 3) & 31, tile = i >> 8;
    int rt = tile >> 3, ct = tile & 7;
    int orow = rt * 16 + ((ln >> 4) << 3) + r;
    int col  = ct * 16 + (ln & 15);
    int b = orow / 36, pix = orow - b * 36;
    int src = (b * 128 + col) * 36 + pix;
    s2s[i]   = (_Float16)c2s[src];
    cur2s[i] = c2s[147456 + src];
    vol2s[i] = c2s[294912 + src];
  }
  for (int i = tid; i < 8192; i += 1024) {        // [32][256] states
    sp3[i] = (_Float16)c3s[i];  cur3[i] = c3s[8192 + i];  vol3[i] = c3s[16384 + i];
    sp4[i] = (_Float16)tcs[i];  cur4[i] = tcs[8192 + i];  vol4[i] = tcs[16384 + i];
    sp5[i] = (_Float16)rsin[i]; cur5[i] = rsin[8192 + i]; vol5[i] = rsin[16384 + i];
  }
  for (int i = tid; i < 4096; i += 1024) {        // [32][128]
    sp6[i] = (_Float16)fsin[i]; cur6[i] = fsin[4096 + i]; vol6[i] = fsin[8192 + i];
  }
  for (int i = tid; i < 24576; i += 1024) ins[i] = (_Float16)0.f;   // ring buf0 = 0
  for (int i = tid; i < 12800; i += 1024) lds_xstage[i] = 0.f;      // writer for compiler
  for (int i = tid; i < 2048; i += 1024) {        // lds w1t[k][n], k padded to 32
    int k = i >> 6, n = i & 63;
    lds_w1t[i] = (k < 9) ? (_Float16)w1[n * 9 + k] : (_Float16)0.f;
  }
  if (tid < 64)  { lds_b1[tid] = b1c[tid]; lds_acc[tid] = 0.f; }
  if (tid < 128) { lds_b2[tid] = b2c[tid]; lds_bf1[tid] = f1bc[tid]; }
  if (tid < 256) { lds_b3[tid] = b3c[tid]; lds_brec[tid] = recbc[tid]; lds_fc2w[tid] = f2w[tid]; }
  if (tid < 768) { lds_btc[tid] = tcbc[tid]; }
  __threadfence_block();
  __syncthreads();

  // ============================ time loop ============================
  for (int t = 0; t < 100; ++t) {
    // ---- TDM: DMA 4 timestep columns of x ([3200][100] f32) into LDS ----
    if ((t & 3) == 0) {
      if (wv == 0) {
        unsigned long long ga = (unsigned long long)(uintptr_t)x +
                                (unsigned long long)t * 4ull;
        unsigned int ldsa = (unsigned int)(uintptr_t)&lds_xstage[0];
        u32x4 g0 = { 1u,                                   // count=1
                     ldsa,                                 // lds_addr
                     (unsigned int)ga,                     // global_addr[31:0]
                     (unsigned int)((ga >> 32) & 0x01FFFFFFull) | 0x80000000u }; // [56:32]|type=2
        i32x8 g1 = { 0x20000,                // data_size=2 (4B)
                     (int)(100u << 16),      // tensor_dim0 = 100
                     (int)(3200u << 16),     // tensor_dim1 = 3200 (low16)
                     (int)(4u << 16),        // tile_dim0 = 4
                     3200,                   // tile_dim1 = 3200
                     100,                    // tensor_dim0_stride = 100
                     0, 0 };
        i32x4 gz = { 0, 0, 0, 0 };
#if __clang_major__ <= 22
        __builtin_amdgcn_tensor_load_to_lds(g0, g1, gz, gz, 0);
#else
        i32x8 gz8 = { 0, 0, 0, 0, 0, 0, 0, 0 };
        __builtin_amdgcn_tensor_load_to_lds(g0, g1, gz, gz, gz8, 0);
#endif
        __builtin_amdgcn_s_wait_tensorcnt(0);
      }
      __syncthreads();
    }
    // ---- convert staged column to f16 ----
    {
      const int sub = t & 3;
      for (int f = tid; f < 3200; f += 1024)
        lds_x[f] = (_Float16)lds_xstage[f * 4 + sub];
    }
    __syncthreads();

    // ---- conv1 (GEMM 2048x64, K=9 padded to 32) + LIF1 ----
    {
      const int ct = wv & 3;
      const int rt0 = (wv >> 2) << 4;
      const int col = ct * 16 + laneM;
      const float bias = lds_b1[col];
      V16 Bv;
      { const _Float16* bp = &lds_w1t[h8 * 64 + col];
        for (int s = 0; s < 8; ++s) { Bv.v[s] = bp[s * 64]; Bv.v[8 + s] = bp[(16 + s) * 64]; } }
      for (int it = 0; it < 16; ++it) {
        int rt = rt0 + it;
        int row = rt * 16 + laneM;
        int b = row >> 6, pix = row & 63, i0 = pix >> 3, j0 = pix & 7;
        const _Float16* xb = &lds_x[b * 100 + i0 * 10 + j0];
        V16 A = {};
        if (lane < 16) {                 // K = 0..7 taps
          A.v[0] = xb[0];  A.v[1] = xb[1];  A.v[2] = xb[2];
          A.v[3] = xb[10]; A.v[4] = xb[11]; A.v[5] = xb[12];
          A.v[6] = xb[20]; A.v[7] = xb[21];
        } else {                         // K = 8 only
          A.v[0] = xb[22];
        }
        v8f acc = {};
        WMMA16(acc, A.v, Bv.v);
        const size_t base = ((size_t)((rt * 4 + ct) * 32 + lane)) << 3;
        v8f curv = *(const v8f*)(cur1s + base);
        v8f volv = *(const v8f*)(vol1s + base);
        const v8h spv = *(const v8h*)(s1s + base);
        v8h snew;
        #pragma unroll
        for (int r = 0; r < 8; ++r) {
          float c = __builtin_fmaf(0.5f, curv[r], acc[r] + bias);
          float sprev = (float)spv[r];
          float v = __builtin_fmaf(0.75f * (1.0f - sprev), volv[r], c);
          float s = (v > 0.5f) ? 1.0f : 0.0f;
          curv[r] = c; volv[r] = v; snew[r] = (_Float16)s;
        }
        *(v8f*)(cur1s + base) = curv;
        *(v8f*)(vol1s + base) = volv;
        *(v8h*)(s1s + base) = snew;
        #pragma unroll
        for (int r = 0; r < 8; ++r)       // NHWC scatter for conv2 A
          s1[((size_t)(rt * 16 + h8 + r) << 6) + col] = snew[r];
      }
    }
    __threadfence_block(); __syncthreads();

    // ---- conv2 (implicit GEMM 1152x128, K=576) + LIF2 ----
    {
      const int ct = wv & 7;
      const int rt0 = (wv >> 3) * 18;
      const int col = ct * 16 + laneM;
      const float bias = lds_b2[col];
      for (int it = 0; it < 18; ++it) {
        int rt = rt0 + it;
        int rowA = rt * 16 + laneM;
        int b = rowA / 36, p = rowA - b * 36;
        int i0 = p / 6, j0 = p - i0 * 6;
        v8f acc = {};
        for (int kt = 0; kt < 18; ++kt) {
          int khw = kt >> 1, cb = (kt & 1) << 5;
          int ki = khw / 3, kj = khw - ki * 3;
          const _Float16* ap = s1 + (((size_t)(b << 6) + (i0 + ki) * 8 + (j0 + kj)) << 6) + cb + h8;
          const _Float16* bp = w2t + (((size_t)khw * 64 + cb + h8) << 7) + col;
          v16h A = ld_pair(ap, 16);
          v16h B; for (int s = 0; s < 8; ++s) { B[s] = bp[s << 7]; B[8 + s] = bp[(16 + s) << 7]; }
          WMMA16(acc, A, B);
        }
        const size_t base = ((size_t)((rt * 8 + ct) * 32 + lane)) << 3;
        v8f curv = *(const v8f*)(cur2s + base);
        v8f volv = *(const v8f*)(vol2s + base);
        const v8h spv = *(const v8h*)(s2s + base);
        v8h snew;
        #pragma unroll
        for (int r = 0; r < 8; ++r) {
          float c = __builtin_fmaf(0.5f, curv[r], acc[r] + bias);
          float sprev = (float)spv[r];
          float v = __builtin_fmaf(0.75f * (1.0f - sprev), volv[r], c);
          float s = (v > 0.5f) ? 1.0f : 0.0f;
          curv[r] = c; volv[r] = v; snew[r] = (_Float16)s;
        }
        *(v8f*)(cur2s + base) = curv;
        *(v8f*)(vol2s + base) = volv;
        *(v8h*)(s2s + base) = snew;
        #pragma unroll
        for (int r = 0; r < 8; ++r)       // NHWC scatter for pooling
          s2[((size_t)(rt * 16 + h8 + r) << 7) + col] = snew[r];
      }
    }
    __threadfence_block(); __syncthreads();

    // ---- avgpool 2x2 (6x6->3x3, NHWC), 8 channels per iter, pk-f16 math ----
    for (int u = tid; u < 4608; u += 1024) {
      int cc = (u & 15) << 3;            // channel block base
      int rest = u >> 4;                 // b*9 + (pi*3+pj)
      int b = rest / 9, pp = rest - b * 9;
      int pi = pp / 3, pj = pp - pi * 3;
      int r00 = b * 36 + pi * 12 + pj * 2;
      const _Float16* sp = s2 + ((size_t)r00 << 7) + cc;
      v8h a0 = *(const v8h*)sp;
      v8h a1 = *(const v8h*)(sp + 128);
      v8h a2 = *(const v8h*)(sp + 6 * 128);
      v8h a3 = *(const v8h*)(sp + 7 * 128);
      v8h sum = (a0 + a1) + (a2 + a3);
      v8h res;
      #pragma unroll
      for (int r = 0; r < 8; ++r) res[r] = sum[r] * (_Float16)0.25f;
      *(v8h*)(p2 + ((size_t)rest << 7) + cc) = res;
    }
    __threadfence_block(); __syncthreads();

    // ---- conv3 (GEMM 32x256, K=1152) + LIF3 -> ring slot t%3 ----
    {
      const int rt = wv >> 4, ct = wv & 15;
      const int bA = rt * 16 + laneM;
      const int d  = ct * 16 + laneM;
      v8f acc = {};
      for (int kt = 0; kt < 36; ++kt) {
        int khw = kt >> 2, cb = (kt & 3) << 5;
        const _Float16* ap = p2 + (((size_t)bA * 9 + khw) << 7) + cb + h8;
        const _Float16* bp = w3t + (((size_t)khw * 128 + cb + h8) << 8) + d;
        v16h A = ld_pair(ap, 16);
        v16h B; for (int s = 0; s < 8; ++s) { B[s] = bp[s << 8]; B[8 + s] = bp[(16 + s) << 8]; }
        WMMA16(acc, A, B);
      }
      _Float16* insNew = ins + (t % 3) * 8192;
      const float bias = lds_b3[d];
      #pragma unroll
      for (int r = 0; r < 8; ++r) {
        int b = rt * 16 + h8 + r;
        size_t idx = ((size_t)b << 8) + d;
        float s = lif_step(acc[r] + bias, cur3 + idx, vol3 + idx, sp3 + idx);
        insNew[idx] = (_Float16)s;
      }
    }
    __threadfence_block(); __syncthreads();

    // ---- temporal conv (GEMM 32x256, K=768 over ring) + LIF4 ----
    {
      const int rt = wv >> 4, ct = wv & 15;
      const int bA = rt * 16 + laneM;
      const int d  = ct * 16 + laneM;
      v8f acc = {};
      for (int kt = 0; kt < 24; ++kt) {
        int k = kt >> 3, cb = (kt & 7) << 5;
        int slot = (t + 1 + k) % 3;                       // oldest..newest
        const _Float16* ap = ins + slot * 8192 + ((size_t)bA << 8) + cb + h8;
        const _Float16* bp = wtct + (((size_t)k * 256 + cb + h8) << 8) + d;
        v16h A = ld_pair(ap, 16);
        v16h B; for (int s = 0; s < 8; ++s) { B[s] = bp[s << 8]; B[8 + s] = bp[(16 + s) << 8]; }
        WMMA16(acc, A, B);
      }
      float bias = lds_btc[512 + d];                      // k=2 always valid
      if (t >= 1) bias += lds_btc[256 + d];
      if (t >= 2) bias += lds_btc[d];
      #pragma unroll
      for (int r = 0; r < 8; ++r) {
        int b = rt * 16 + h8 + r;
        size_t idx = ((size_t)b << 8) + d;
        lif_step(acc[r] + bias, cur4 + idx, vol4 + idx, sp4 + idx);
      }
    }
    __threadfence_block(); __syncthreads();

    // ---- recurrent (GEMM 32x256 on prev sp5) + identity psp + LIF5 ----
    {
      const int rt = wv >> 4, ct = wv & 15;
      const int bA = rt * 16 + laneM;
      const int d  = ct * 16 + laneM;
      v8f acc = {};
      for (int kt = 0; kt < 8; ++kt) {
        int cb = kt << 5;
        const _Float16* ap = sp5 + ((size_t)bA << 8) + cb + h8;
        const _Float16* bp = wrect + (((size_t)(cb + h8)) << 8) + d;
        v16h A = ld_pair(ap, 16);
        v16h B; for (int s = 0; s < 8; ++s) { B[s] = bp[s << 8]; B[8 + s] = bp[(16 + s) << 8]; }
        WMMA16(acc, A, B);
      }
      __syncthreads();   // all reads of prev sp5 complete before overwrite
      const float bias = lds_brec[d];
      #pragma unroll
      for (int r = 0; r < 8; ++r) {
        int b = rt * 16 + h8 + r;
        size_t idx = ((size_t)b << 8) + d;
        float psp = acc[r] + (float)sp4[idx] + bias;
        lif_step(psp, cur5 + idx, vol5 + idx, sp5 + idx);
      }
    }
    __threadfence_block(); __syncthreads();

    // ---- fc1 (GEMM 32x128, K=256) + dropout-masked LIF6 (waves 0-15) ----
    if (wv < 16) {
      const int rt = wv >> 3, ct = wv & 7;
      const int bA = rt * 16 + laneM;
      const int n  = ct * 16 + laneM;
      v8f acc = {};
      for (int kt = 0; kt < 8; ++kt) {
        int cb = kt << 5;
        const _Float16* ap = sp5 + ((size_t)bA << 8) + cb + h8;
        const _Float16* bp = wf1t + (((size_t)(cb + h8)) << 7) + n;
        v16h A = ld_pair(ap, 16);
        v16h B; for (int s = 0; s < 8; ++s) { B[s] = bp[s << 7]; B[8 + s] = bp[(16 + s) << 7]; }
        WMMA16(acc, A, B);
      }
      const float bias = lds_bf1[n];
      #pragma unroll
      for (int r = 0; r < 8; ++r) {
        int b = rt * 16 + h8 + r;
        size_t idx = ((size_t)b << 7) + n;
        float c6 = __builtin_fmaf(0.5f, cur6[idx], acc[r] + bias) * maskfc[idx];
        float sprev = (float)sp6[idx];
        float v6 = __builtin_fmaf(0.75f * (1.0f - sprev), vol6[idx], c6);
        float s = (v6 > 0.5f) ? 1.0f : 0.0f;
        cur6[idx] = c6; vol6[idx] = v6; sp6[idx] = (_Float16)s;
      }
    }
    __threadfence_block(); __syncthreads();

    // ---- fc2 + weighted time accumulation ----
    if (tid < 64) {
      int b = tid >> 1, o = tid & 1;
      const _Float16* sp = sp6 + ((size_t)b << 7);
      const float* wrow = &lds_fc2w[o * 128];
      float sum = 0.f;
      for (int f = 0; f < 128; ++f) sum = __builtin_fmaf((float)sp[f], wrow[f], sum);
      lds_acc[tid] += sum * tswp[t];
    }
    __syncthreads();
  }

  if (tid < 64) out[tid] = lds_acc[tid];
}

extern "C" void kernel_launch(void* const* d_in, const int* in_sizes, int n_in,
                              void* d_out, int out_size, void* d_ws, size_t ws_size,
                              hipStream_t stream) {
  (void)in_sizes; (void)n_in; (void)out_size; (void)ws_size;
  snn_fused<<<dim3(1), dim3(1024), 0, stream>>>(
      (const float*)d_in[0],  (const float*)d_in[1],  (const float*)d_in[2],
      (const float*)d_in[3],  (const float*)d_in[4],  (const float*)d_in[5],
      (const float*)d_in[6],  (const float*)d_in[7],  (const float*)d_in[8],
      (const float*)d_in[9],  (const float*)d_in[10], (const float*)d_in[11],
      (const float*)d_in[12], (const float*)d_in[13], (const float*)d_in[14],
      (const float*)d_in[15], (const float*)d_in[16], (const float*)d_in[17],
      (const float*)d_in[18], (const float*)d_in[19], (const float*)d_in[20],
      (const float*)d_in[21], (float*)d_out, (char*)d_ws);
}
